// VideoCharBiLSTMEmbedder_31542239821978
// MI455X (gfx1250) — matrix operand order repaired and checked by reading, not compiled, not run on hardware
//
#include <hip/hip_runtime.h>
#include <hip/hip_bf16.h>

typedef __attribute__((ext_vector_type(16))) _Float16 v16h;
typedef __attribute__((ext_vector_type(8)))  float    v8f;
typedef __attribute__((ext_vector_type(2)))  __fp16   v2hf;   // matches cvt_pkrtz return type

#define N_WORDS_C 100000
#define MAX_LEN_C 24
#define EMB_C 50
#define HID_C 50
#define KPAD 64            // padded inner dim; col 50 of Wih carries bias, x[50]==1
#define EMB_ROWS 200
#define WS_EMB_OFF 0
#define WS_FRAG_OFF (EMB_ROWS*KPAD*2)          // 25600 B, 16B aligned
#define FRAG_BYTES 1024                         // 32 lanes * 32 B
#define FRAGS_PER_DIR 64                        // which(2) * tile(16) * kc(2)
#define DIR_FRAG_BYTES (FRAGS_PER_DIR*FRAG_BYTES)   // 65536 B

// ---------- pack kernels (tiny, run on stream before main kernel) ----------

// embedding table -> f16, padded K=64, x[50]=1 (bias lane), x[51..63]=0
__global__ void pack_emb_kernel(const float* __restrict__ emb, _Float16* __restrict__ dst) {
    int id = blockIdx.x * blockDim.x + threadIdx.x;
    if (id >= EMB_ROWS * KPAD) return;
    int v = id >> 6, k = id & 63;
    float val = (k < EMB_C) ? emb[v * EMB_C + k] : (k == EMB_C ? 1.0f : 0.0f);
    dst[id] = (_Float16)val;
}

// Wih/Whh/b -> WMMA A-fragment layout in workspace.
// A-frag (16x32 f16): lane l holds m = l&15; half e: K = 16*(l>>4) + e (+32*kc)
// padded gate row m: blk = m>>6 (i/f/g/o), kk = m&63; kk>=50 rows are zero.
__global__ void pack_wfrag_kernel(const float* __restrict__ Wih_f, const float* __restrict__ Whh_f,
                                  const float* __restrict__ b_f,
                                  const float* __restrict__ Wih_b, const float* __restrict__ Whh_b,
                                  const float* __restrict__ b_b,
                                  _Float16* __restrict__ dst) {
    int tid = blockIdx.x * blockDim.x + threadIdx.x;   // 0..4095
    if (tid >= 128 * 32) return;
    int lane = tid & 31;
    int frag = tid >> 5;            // 0..127
    int kc    =  frag        & 1;
    int tile  = (frag >> 1)  & 15;
    int which = (frag >> 5)  & 1;   // 0 = ih (+bias col), 1 = hh
    int dir   =  frag >> 6;         // 0 = fwd, 1 = bwd
    const float* W = which ? (dir ? Whh_b : Whh_f) : (dir ? Wih_b : Wih_f);
    const float* B = dir ? b_b : b_f;
    int m16 = lane & 15, hi = lane >> 4;
    _Float16* out = dst + frag * (FRAG_BYTES / 2) + lane * 16;
#pragma unroll
    for (int e = 0; e < 16; ++e) {
        int gm = tile * 16 + m16;                       // padded gate row 0..255
        int K  = 32 * kc + 16 * hi + e;
        int blk = gm >> 6, kk = gm & 63;
        float val = 0.0f;
        if (kk < HID_C) {
            int row = blk * HID_C + kk;                 // row in original 200-row weight
            if (which == 0) {
                if (K < EMB_C)       val = W[row * EMB_C + K];
                else if (K == EMB_C) val = B[row];      // bias folded into col 50
            } else {
                if (K < HID_C)       val = W[row * HID_C + K];
            }
        }
        out[e] = (_Float16)val;
    }
}

// ---------- main biLSTM kernel ----------

__device__ __forceinline__ float sigm_f(float x) {
    return __builtin_amdgcn_rcpf(1.0f + __expf(-x));
}
__device__ __forceinline__ float tanh_f(float x) {
    float e = __expf(2.0f * x);
    return 1.0f - 2.0f * __builtin_amdgcn_rcpf(e + 1.0f);
}

__global__ void __launch_bounds__(256)
bilstm_wmma_kernel(const int* __restrict__ ci, const int* __restrict__ lens,
                   const _Float16* __restrict__ embp, const _Float16* __restrict__ wfrag,
                   float* __restrict__ out) {
    extern __shared__ __align__(32) char smem[];
    const int tid = threadIdx.x;
    const int dir = blockIdx.y;

    // Cooperative copy: this direction's packed weight fragments (64KB) -> LDS
    {
        const uint4* src = (const uint4*)(wfrag + dir * (DIR_FRAG_BYTES / 2));
        uint4* dst = (uint4*)smem;
#pragma unroll
        for (int i = 0; i < 16; ++i) dst[tid + i * 256] = src[tid + i * 256];
    }
    __syncthreads();

    const int lane = tid & 31;
    const int wv   = tid >> 5;                  // 8 waves, 16 words each
    const int hi   = lane >> 4;                 // lane half
    const int word = blockIdx.x * 128 + wv * 16 + (lane & 15);
    const bool valid = word < N_WORDS_C;
    const int  lenB  = valid ? lens[word] : 0;
    const int  cibase = valid ? word * MAX_LEN_C : 0;

    // State: c in f32 (D layout), h as packed f16 pairs (k = 16j + 8*hi + 2s (+1))
    float    c[4][8];
    unsigned hpk[4][4];
#pragma unroll
    for (int j = 0; j < 4; ++j) {
#pragma unroll
        for (int r = 0; r < 8; ++r) c[j][r] = 0.0f;
#pragma unroll
        for (int s = 0; s < 4; ++s) hpk[j][s] = 0u;
    }

#pragma unroll 1
    for (int t = 0; t < MAX_LEN_C; ++t) {
        int tpos;
        if (dir == 0) {
            tpos = t;
        } else {
            int p = lenB - 1 - t;
            p = p < 0 ? 0 : p;
            p = p > (MAX_LEN_C - 1) ? (MAX_LEN_C - 1) : p;
            tpos = p;
        }
        int idx = ci[cibase + tpos];

        // x as B operand: lane n = l&15 (word), halves e: K = 32*kc + 16*hi + e -> contiguous
        const _Float16* xr = embp + idx * KPAD + hi * 16;
        v16h xb0 = *(const v16h*)(xr);
        v16h xb1 = *(const v16h*)(xr + 32);

        // Build h B-fragments from packed h via lane-half swap (xor-16 shuffle) + select
        v16h hb[2];
#pragma unroll
        for (int kc = 0; kc < 2; ++kc) {
            union { unsigned u[8]; v16h v; } hu;
#pragma unroll
            for (int d = 0; d < 8; ++d) {
                int s = d & 3;
                unsigned lo  = hpk[2 * kc][s];
                unsigned hh2 = hpk[2 * kc + 1][s];
                if (d < 4) {
                    unsigned sw = (unsigned)__shfl_xor((int)hh2, 16);
                    hu.u[d] = hi ? sw : lo;
                } else {
                    unsigned sw = (unsigned)__shfl_xor((int)lo, 16);
                    hu.u[d] = hi ? hh2 : sw;
                }
            }
            hb[kc] = hu.v;
        }

        const bool mk = t < lenB;

        // Gate tiles grouped so i/f/g/o for hidden chunk j are tiles j, j+4, j+8, j+12
#pragma unroll
        for (int j = 0; j < 4; ++j) {
            // Launder a zero offset through an opaque asm so the LDS fragment loads
            // below cannot be hoisted out of the t-loop (which previously forced
            // ~512 VGPRs of weights live -> scratch spills). Loads stay ds_load_*
            // because address-space inference still sees `smem + offset`.
            unsigned zoff = 0;
            asm volatile("" : "+v"(zoff));
            const char* sb = smem + zoff + lane * 32;

            v8f acc[4];
#pragma unroll
            for (int gb = 0; gb < 4; ++gb) {
                const int tile = gb * 4 + j;
                const char* ih0 = sb + ((0 * 16 + tile) * 2 + 0) * FRAG_BYTES;
                const char* ih1 = sb + ((0 * 16 + tile) * 2 + 1) * FRAG_BYTES;
                const char* hh0 = sb + ((1 * 16 + tile) * 2 + 0) * FRAG_BYTES;
                const char* hh1 = sb + ((1 * 16 + tile) * 2 + 1) * FRAG_BYTES;
                v8f a = {0.f, 0.f, 0.f, 0.f, 0.f, 0.f, 0.f, 0.f};
                a = __builtin_amdgcn_wmma_f32_16x16x32_f16(false, *(const v16h*)ih0,
                                                           false, xb0, (short)0, a, false, false);
                a = __builtin_amdgcn_wmma_f32_16x16x32_f16(false, *(const v16h*)ih1,
                                                           false, xb1, (short)0, a, false, false);
                a = __builtin_amdgcn_wmma_f32_16x16x32_f16(false, *(const v16h*)hh0,
                                                           false, hb[0], (short)0, a, false, false);
                a = __builtin_amdgcn_wmma_f32_16x16x32_f16(false, *(const v16h*)hh1,
                                                           false, hb[1], (short)0, a, false, false);
                acc[gb] = a;
            }
            float hn[8];
#pragma unroll
            for (int r = 0; r < 8; ++r) {
                float i_ = sigm_f(acc[0][r]);
                float f_ = sigm_f(acc[1][r]);
                float g_ = tanh_f(acc[2][r]);
                float o_ = sigm_f(acc[3][r]);
                float cn = f_ * c[j][r] + i_ * g_;
                hn[r] = o_ * tanh_f(cn);
                c[j][r] = mk ? cn : c[j][r];
            }
#pragma unroll
            for (int s = 0; s < 4; ++s) {
                v2hf p = __builtin_amdgcn_cvt_pkrtz(hn[2 * s], hn[2 * s + 1]);
                unsigned pu = __builtin_bit_cast(unsigned, p);
                hpk[j][s] = mk ? pu : hpk[j][s];
            }
        }
    }

    // Output: h[k] for k<50, concat [fwd | bwd]
    if (valid) {
        float* o = out + word * (2 * HID_C) + dir * HID_C;
#pragma unroll
        for (int j = 0; j < 4; ++j) {
#pragma unroll
            for (int s = 0; s < 4; ++s) {
                v2hf p = __builtin_bit_cast(v2hf, hpk[j][s]);
                int k0 = 16 * j + 8 * hi + 2 * s;
                if (k0 < HID_C)     o[k0]     = (float)p.x;
                if (k0 + 1 < HID_C) o[k0 + 1] = (float)p.y;
            }
        }
    }
}

// ---------- launcher ----------

extern "C" void kernel_launch(void* const* d_in, const int* in_sizes, int n_in,
                              void* d_out, int out_size, void* d_ws, size_t ws_size,
                              hipStream_t stream) {
    const int*   ci    = (const int*)d_in[0];
    const int*   lens  = (const int*)d_in[1];
    const float* emb   = (const float*)d_in[2];
    const float* Wih_f = (const float*)d_in[3];
    const float* Whh_f = (const float*)d_in[4];
    const float* b_f   = (const float*)d_in[5];
    const float* Wih_b = (const float*)d_in[6];
    const float* Whh_b = (const float*)d_in[7];
    const float* b_b   = (const float*)d_in[8];

    _Float16* embp  = (_Float16*)((char*)d_ws + WS_EMB_OFF);
    _Float16* wfrag = (_Float16*)((char*)d_ws + WS_FRAG_OFF);

    pack_emb_kernel<<<(EMB_ROWS * KPAD + 255) / 256, 256, 0, stream>>>(emb, embp);
    pack_wfrag_kernel<<<16, 256, 0, stream>>>(Wih_f, Whh_f, b_f, Wih_b, Whh_b, b_b, wfrag);

    dim3 grid((N_WORDS_C + 127) / 128, 2);
    bilstm_wmma_kernel<<<grid, 256, DIR_FRAG_BYTES, stream>>>(ci, lens, embp, wfrag, (float*)d_out);
}
